// GraphD_28372553957675
// MI455X (gfx1250) — matrix operand order repaired and checked by reading, not compiled, not run on hardware
//
#include <hip/hip_runtime.h>

typedef __attribute__((ext_vector_type(16))) _Float16 v16h;
typedef __attribute__((ext_vector_type(8)))  _Float16 v8h;
typedef __attribute__((ext_vector_type(8)))  float    v8f;

#define NN 50000
#define NE 800000
#define DOUT 512   // Wq|Wk|Wv|Wskip concatenated, each 128 cols
#define NB_SCAN 196  // ceil(NN/256)

// ---------------- feature f32 -> f16 ----------------
__global__ void cvt_f32_to_f16(const float* __restrict__ src,
                               _Float16* __restrict__ dst, int n) {
  int i = blockIdx.x * blockDim.x + threadIdx.x;
  if (i < n) dst[i] = (_Float16)src[i];
}

// ---------------- pack weights into WMMA B-operand layout ----------------
// Wpack linear index t = kb*16384 + ct*512 + lane*16 + j  maps to
//   column n = ct*16 + (lane&15), K = kb*32 + (lane>=16 ? 16:0) + j
__global__ void pack_weights(const float* __restrict__ Wq, const float* __restrict__ Wk,
                             const float* __restrict__ Wv, const float* __restrict__ Ws,
                             const float* __restrict__ bq, const float* __restrict__ bk,
                             const float* __restrict__ bv, const float* __restrict__ bs,
                             _Float16* __restrict__ Wpack, float* __restrict__ bcat,
                             int din) {
  int t = blockIdx.x * blockDim.x + threadIdx.x;
  int total = din * DOUT;
  if (t >= total) return;
  int j    = t & 15;
  int lane = (t >> 4) & 31;
  int ct   = (t >> 9) & 31;
  int kb   = t >> 14;
  int n    = ct * 16 + (lane & 15);
  int seg  = n >> 7;
  int col  = n & 127;
  int k    = kb * 32 + ((lane & 16) ? 16 : 0) + j;
  const float* W = (seg == 0) ? Wq : (seg == 1) ? Wk : (seg == 2) ? Wv : Ws;
  Wpack[t] = (_Float16)W[(size_t)k * 128 + col];
  if (t < DOUT) {
    const float* B = (t < 128) ? bq : (t < 256) ? bk : (t < 384) ? bv : bs;
    bcat[t] = B[t & 127];
  }
}

// ---------------- WMMA GEMM: [NN, din] x [din, 512] + bias -> QKVS ----------------
__global__ void __launch_bounds__(256)
gemm_wmma(const _Float16* __restrict__ A, const _Float16* __restrict__ Wpack,
          const float* __restrict__ bcat, float* __restrict__ Cmat, int din) {
  const int lane    = threadIdx.x & 31;
  const int wave    = (blockIdx.x * blockDim.x + threadIdx.x) >> 5;
  const int ct      = wave & 31;   // 32 column tiles (512/16)
  const int rt      = wave >> 5;   // 3125 row tiles (50000/16)
  const int halfSel = (lane & 16) ? 1 : 0;
  const int row     = rt * 16 + (lane & 15);
  const int nkb     = din >> 5;

  const _Float16* arow = A + (size_t)row * din + (halfSel ? 16 : 0);
  const _Float16* bptr = Wpack + (size_t)ct * 512 + (size_t)lane * 16;

  v8f acc = {};
  for (int kb = 0; kb < nkb; ++kb) {
    v8h a0 = *(const v8h*)(arow + kb * 32);        // K 0..7  (or 16..23)
    v8h a1 = *(const v8h*)(arow + kb * 32 + 8);    // K 8..15 (or 24..31)
    v16h a = __builtin_shufflevector(a0, a1, 0, 1, 2, 3, 4, 5, 6, 7,
                                     8, 9, 10, 11, 12, 13, 14, 15);
    v16h b = *(const v16h*)(bptr + (size_t)kb * 16384);
    acc = __builtin_amdgcn_wmma_f32_16x16x32_f16(false, a, false, b,
                                                 (short)0, acc, false, false);
  }
  const int   col  = ct * 16 + (lane & 15);
  const float bias = bcat[col];
  float* out = Cmat + (size_t)(rt * 16 + (halfSel ? 8 : 0)) * DOUT + col;
#pragma unroll
  for (int r = 0; r < 8; ++r) out[(size_t)r * DOUT] = acc[r] + bias;
}

// ================= CSR build (once per call) =================
__global__ void zero_int(int* __restrict__ p, int n) {
  int i = blockIdx.x * blockDim.x + threadIdx.x;
  if (i < n) p[i] = 0;
}

__global__ void hist_dst(const int* __restrict__ ei, int* __restrict__ deg) {
  int e = blockIdx.x * blockDim.x + threadIdx.x;
  if (e < NE) atomicAdd(&deg[ei[NE + e]], 1);
}

// per-block exclusive scan of deg -> rowptr (block-local), block totals -> bsum
__global__ void scan_block(const int* __restrict__ deg, int* __restrict__ rowptr,
                           int* __restrict__ bsum) {
  __shared__ int sm[256];
  int i = blockIdx.x * 256 + threadIdx.x;
  int v = (i < NN) ? deg[i] : 0;
  sm[threadIdx.x] = v;
  __syncthreads();
  for (int off = 1; off < 256; off <<= 1) {
    int x = (threadIdx.x >= (unsigned)off) ? sm[threadIdx.x - off] : 0;
    __syncthreads();
    sm[threadIdx.x] += x;
    __syncthreads();
  }
  if (i < NN) rowptr[i] = sm[threadIdx.x] - v;  // exclusive within block
  if (threadIdx.x == 255) bsum[blockIdx.x] = sm[255];
}

// exclusive scan of block sums (single block; NB_SCAN <= 256)
__global__ void scan_bsum(int* __restrict__ bsum, int nb) {
  __shared__ int sm[256];
  int t = threadIdx.x;
  int v = (t < nb) ? bsum[t] : 0;
  sm[t] = v;
  __syncthreads();
  for (int off = 1; off < 256; off <<= 1) {
    int x = (t >= off) ? sm[t - off] : 0;
    __syncthreads();
    sm[t] += x;
    __syncthreads();
  }
  if (t < nb) bsum[t] = sm[t] - v;
}

__global__ void scan_finish(int* __restrict__ rowptr, const int* __restrict__ bsum,
                            int* __restrict__ cursor) {
  int i = blockIdx.x * blockDim.x + threadIdx.x;
  if (i < NN) {
    int r = rowptr[i] + bsum[i >> 8];
    rowptr[i] = r;
    cursor[i] = r;
  }
}

__global__ void reorder_edges(const int* __restrict__ ei, int* __restrict__ cursor,
                              int* __restrict__ esrc) {
  int e = blockIdx.x * blockDim.x + threadIdx.x;
  if (e < NE) {
    int pos = atomicAdd(&cursor[ei[NE + e]], 1);
    esrc[pos] = ei[e];
  }
}

// ========== fused per-(node,head) attention: logits -> softmax -> agg -> skip/ReLU ==========
__global__ void __launch_bounds__(256)
node_attn(const int* __restrict__ rowptr, const int* __restrict__ deg,
          const int* __restrict__ esrc, const float* __restrict__ QKVS,
          float* __restrict__ logits, float* __restrict__ out, int relu) {
  int t = blockIdx.x * blockDim.x + threadIdx.x;
  if (t >= NN * 4) return;
  int n = t >> 2, h = t & 3;
  int start = rowptr[n], dg = deg[n];

  const float4* qp = (const float4*)(QKVS + (size_t)n * DOUT + h * 32);  // Q segment
  float4 q[8];
#pragma unroll
  for (int j = 0; j < 8; ++j) q[j] = qp[j];

  // pass 1: logits + running max (logits spilled to L2-resident scratch)
  float m = -3.0e38f;
  for (int i = 0; i < dg; ++i) {
    int src = esrc[start + i];
    const float4* kp = (const float4*)(QKVS + (size_t)src * DOUT + 128 + h * 32);  // K
    float s = 0.0f;
#pragma unroll
    for (int j = 0; j < 8; ++j) {
      float4 k = kp[j];
      s += q[j].x * k.x + q[j].y * k.y + q[j].z * k.z + q[j].w * k.w;
    }
    s *= 0.17677669529663687f;  // 1/sqrt(32)
    logits[(size_t)(start + i) * 4 + h] = s;
    m = fmaxf(m, s);
  }

  // pass 2: exp/sum + weighted V accumulation in registers (no atomics)
  float4 acc[8];
#pragma unroll
  for (int j = 0; j < 8; ++j) acc[j] = make_float4(0.f, 0.f, 0.f, 0.f);
  float ssum = 0.0f;
  for (int i = 0; i < dg; ++i) {
    int src = esrc[start + i];
    float w = __expf(logits[(size_t)(start + i) * 4 + h] - m);
    ssum += w;
    const float4* vp = (const float4*)(QKVS + (size_t)src * DOUT + 256 + h * 32);  // V
#pragma unroll
    for (int j = 0; j < 8; ++j) {
      float4 v = vp[j];
      acc[j].x += w * v.x; acc[j].y += w * v.y;
      acc[j].z += w * v.z; acc[j].w += w * v.w;
    }
  }
  float inv = 1.0f / (ssum + 1e-16f);  // isolated nodes: acc==0 -> output = skip only

  const float4* sk = (const float4*)(QKVS + (size_t)n * DOUT + 384 + h * 32);  // skip
  float4* op = (float4*)(out + (size_t)n * 128 + h * 32);
#pragma unroll
  for (int j = 0; j < 8; ++j) {
    float4 s4 = sk[j];
    float4 r;
    r.x = acc[j].x * inv + s4.x;
    r.y = acc[j].y * inv + s4.y;
    r.z = acc[j].z * inv + s4.z;
    r.w = acc[j].w * inv + s4.w;
    if (relu) {
      r.x = fmaxf(r.x, 0.f); r.y = fmaxf(r.y, 0.f);
      r.z = fmaxf(r.z, 0.f); r.w = fmaxf(r.w, 0.f);
    }
    op[j] = r;
  }
}

extern "C" void kernel_launch(void* const* d_in, const int* in_sizes, int n_in,
                              void* d_out, int out_size, void* d_ws, size_t ws_size,
                              hipStream_t stream) {
  (void)in_sizes; (void)n_in; (void)out_size; (void)ws_size;
  const float* x  = (const float*)d_in[0];
  const int*   ei = (const int*)d_in[1];

  char* ws = (char*)d_ws;
  size_t off = 0;
  auto alloc = [&](size_t b) { size_t o = off; off += (b + 255) & ~(size_t)255; return o; };
  _Float16* Hh     = (_Float16*)(ws + alloc((size_t)NN * 128 * 2));
  _Float16* Wp     = (_Float16*)(ws + alloc((size_t)128 * DOUT * 2));
  float*    bcat   = (float*)(ws + alloc((size_t)DOUT * 4));
  float*    QKVS   = (float*)(ws + alloc((size_t)NN * DOUT * 4));
  float*    logits = (float*)(ws + alloc((size_t)NE * 4 * 4));
  int*      deg    = (int*)(ws + alloc((size_t)NN * 4));
  int*      rowptr = (int*)(ws + alloc((size_t)NN * 4));
  int*      bsum   = (int*)(ws + alloc((size_t)NB_SCAN * 4));
  int*      cursor = (int*)(ws + alloc((size_t)NN * 4));
  int*      esrc   = (int*)(ws + alloc((size_t)NE * 4));
  float*    Hbuf   = (float*)(ws + alloc((size_t)NN * 128 * 4));

  // ---- CSR build by destination (once per call) ----
  zero_int<<<(NN + 255) / 256, 256, 0, stream>>>(deg, NN);
  hist_dst<<<(NE + 255) / 256, 256, 0, stream>>>(ei, deg);
  scan_block<<<NB_SCAN, 256, 0, stream>>>(deg, rowptr, bsum);
  scan_bsum<<<1, 256, 0, stream>>>(bsum, NB_SCAN);
  scan_finish<<<(NN + 255) / 256, 256, 0, stream>>>(rowptr, bsum, cursor);
  reorder_edges<<<(NE + 255) / 256, 256, 0, stream>>>(ei, cursor, esrc);

  // ---- 3 TransformerConv layers ----
  for (int li = 0; li < 3; ++li) {
    const int din = (li == 0) ? 64 : 128;
    const float* Hin  = (li == 0) ? x : Hbuf;
    float*       Hout = (li == 2) ? (float*)d_out : Hbuf;
    void* const* p = d_in + 2 + li * 8;  // Wq,bq,Wk,bk,Wv,bv,Wskip,bskip
    const float *Wq  = (const float*)p[0], *bq  = (const float*)p[1];
    const float *Wk  = (const float*)p[2], *bk  = (const float*)p[3];
    const float *Wv  = (const float*)p[4], *bv  = (const float*)p[5];
    const float *Wsk = (const float*)p[6], *bsk = (const float*)p[7];

    const int nh = NN * din;
    cvt_f32_to_f16<<<(nh + 255) / 256, 256, 0, stream>>>(Hin, Hh, nh);

    const int npw = din * DOUT;
    pack_weights<<<(npw + 255) / 256, 256, 0, stream>>>(Wq, Wk, Wv, Wsk,
                                                        bq, bk, bv, bsk, Wp, bcat, din);

    // 3125 row tiles * 32 col tiles = 100000 waves, 8 waves/block
    gemm_wmma<<<(3125 * 32) / 8, 256, 0, stream>>>(Hh, Wp, bcat, QKVS, din);

    const int nt = NN * 4;
    node_attn<<<(nt + 255) / 256, 256, 0, stream>>>(rowptr, deg, esrc, QKVS,
                                                    logits, Hout, (li < 2) ? 1 : 0);
  }
}
